// Architecture_11690900980380
// MI455X (gfx1250) — matrix operand order repaired and checked by reading, not compiled, not run on hardware
//
#include <hip/hip_runtime.h>
#include <math.h>

// ---------------------------------------------------------------------------
// CDNA5 (gfx1250) implicit-GEMM 3x3 conv, v_wmma_f32_16x16x32_bf16.
// Block (4 waves) = one full 64-px image row x (up to) 128 output channels.
// Per wave: 4 M-tiles x NT N-tiles (template) accumulators.
// Weights pre-packed once per launch into fragment-ready bf16 layout.
// ---------------------------------------------------------------------------

typedef __attribute__((ext_vector_type(16))) __bf16 v16bf;
typedef __attribute__((ext_vector_type(8)))  float  v8f;

constexpr int Bc  = 8;
constexpr int Tt  = 8;
constexpr int CIN = 3;
constexpr int Hh  = 64;
constexpr int Ww  = 64;
constexpr int HW  = Hh * Ww;
constexpr int HID = 64;
constexpr int PROCESS_T = Tt + 3 - 1;           // 10

__device__ __forceinline__ unsigned pack_bf2(float a, float b) {
    // fp32 -> bf16 round-to-nearest-even, packed low|high
    unsigned ua = __float_as_uint(a);
    unsigned ub = __float_as_uint(b);
    ua = (ua + 0x7FFFu + ((ua >> 16) & 1u)) >> 16;
    ub = (ub + 0x7FFFu + ((ub >> 16) & 1u)) >> 16;
    return (ua & 0xFFFFu) | (ub << 16);
}

// ---------------------------------------------------------------------------
// Weight pre-pack: fp32 OIHW -> bf16 fragment-ready
// dst dword i = ((tile*nKb + kb)*32 + lane)*8 + j
//   holds W[co = tile*16 + (lane&15), ci(k0)], W[co, ci(k0+1)]
//   k0 = ((lane<16)?0:16) + 2*j ;  kb = cbGlobal*9 + dy*3 + dx
// Channels zero-padded per concat-tensor (C0 block, then C1 block).
// ---------------------------------------------------------------------------
__global__ void pack_weights(const float* __restrict__ src,
                             unsigned* __restrict__ dst,
                             int Cout, int CinTot, int C0, int C1, int total)
{
    const int nCb0 = (C0 + 31) >> 5;
    const int nKb  = (nCb0 + (C1 ? ((C1 + 31) >> 5) : 0)) * 9;
    for (int i = blockIdx.x * 256 + threadIdx.x; i < total;
         i += gridDim.x * 256) {
        int j    = i & 7;
        int lane = (i >> 3) & 31;
        int rest = i >> 8;
        int kb   = rest % nKb;
        int tile = rest / nKb;
        int nn    = lane & 15;
        int kbase = (lane < 16) ? 0 : 16;
        int k0    = kbase + 2 * j;
        int cbG = kb / 9;
        int tap = kb - cbG * 9;
        int dy = tap / 3, dx = tap - dy * 3;
        int sTen = (cbG >= nCb0) ? 1 : 0;
        int cb   = sTen ? (cbG - nCb0) : cbG;
        int Cs   = sTen ? C1 : C0;
        int ciB  = sTen ? C0 : 0;
        int co = tile * 16 + nn;
        int c0 = cb * 32 + k0;
        float f0 = (c0 < Cs)
            ? src[(((long)co * CinTot + ciB + c0) * 3 + dy) * 3 + dx] : 0.f;
        float f1 = (c0 + 1 < Cs)
            ? src[(((long)co * CinTot + ciB + c0 + 1) * 3 + dy) * 3 + dx] : 0.f;
        dst[i] = pack_bf2(f0, f1);
    }
}

// ---------------------------------------------------------------------------
// Fused conv kernel (templated on N-tiles/wave, epilogue mode, chan-guard).
// MODE 0: out = conv + bias
// MODE 1: g = sigmoid(conv+bias); ch<Cout/2 -> g*h (stores r*h), else g (z)
// MODE 2: out = z*h + (1-z)*tanh(conv+bias);  z = zbuf ch 64+co, h = hbuf ch co
// ---------------------------------------------------------------------------
template <int NT, int MODE, bool CG>
__global__ void __launch_bounds__(128)
conv3x3_wmma(const float* __restrict__ in0, int C0, long sb0,
             const float* __restrict__ in1, int C1, long sb1,
             const unsigned* __restrict__ wp, const float* __restrict__ bias,
             int Cout, float* __restrict__ out,
             const float* __restrict__ hbuf, const float* __restrict__ zbuf)
{
    constexpr int ROWD = 18;                    // dword stride per strip pixel
    __shared__ unsigned ldsA[66 * ROWD];        // 66 px x 32 ch bf16 strip

    const int tid  = threadIdx.x;
    const int lane = tid & 31;
    const int wave = tid >> 5;

    // ---- block = one (b, y) image row, 64 pixels ----
    const int b = blockIdx.x / Hh;
    const int y = blockIdx.x - b * Hh;

    const int nCb0 = (C0 + 31) >> 5;
    const int nCb1 = C1 ? ((C1 + 31) >> 5) : 0;
    const int nKb  = (nCb0 + nCb1) * 9;
    const int nTiles = Cout >> 4;

    int  tt[NT];
    bool act[NT];
#pragma unroll
    for (int i = 0; i < NT; ++i) {
        tt[i]  = blockIdx.y * 4 * NT + wave + i * 4;
        act[i] = tt[i] < nTiles;
    }
    const bool anyAct = act[0];                 // tt[0] is the smallest

    const int Mrow = lane & 15;                 // A frag row (ISA 7.12.2)
    const int kbA  = (lane < 16) ? 0 : 16;

    v8f acc[4][NT] = {};                        // [mt][nt]

    for (int s = 0; s < 2; ++s) {
        const float* ptr = s ? in1 : in0;
        const int  Cs    = s ? C1 : C0;
        const long sb    = s ? sb1 : sb0;
        if (Cs == 0 || ptr == nullptr) continue;    // uniform
        const int nCb   = s ? nCb1 : nCb0;
        const int cbOff = s ? nCb0 : 0;

        for (int cb = 0; cb < nCb; ++cb) {
            for (int dy = 0; dy < 3; ++dy) {
                const int yy = y + dy - 1;
                __syncthreads();                // WAR on ldsA
                {   // ---- stage 66px x 32ch strip, branchless guards ----
                    const bool yok = (yy >= 0) && (yy < Hh);
                    const float* rowp = ptr + (long)b * sb + (long)yy * Ww;
                    for (int d = tid; d < 66 * 16; d += 128) {
                        int p  = d >> 4;                    // 0..65
                        int kk = (d & 15) * 2;              // 0..30
                        int xx = p - 1;
                        int c0 = cb * 32 + kk;
                        bool pok = yok && (xx >= 0) && (xx < Ww);
                        bool ok0 = pok, ok1 = pok;
                        if (CG) {                           // only Cin=3 needs it
                            ok0 = pok && (c0 < Cs);
                            ok1 = pok && (c0 + 1 < Cs);
                        }
                        const float* a0 = ok0 ? (rowp + (long)c0 * HW + xx) : ptr;
                        const float* a1 = ok1 ? (rowp + (long)(c0 + 1) * HW + xx) : ptr;
                        float f0 = *a0; f0 = ok0 ? f0 : 0.f;
                        float f1 = *a1; f1 = ok1 ? f1 : 0.f;
                        ldsA[p * ROWD + (kk >> 1)] = pack_bf2(f0, f1);
                    }
                }
                __syncthreads();                // RAW on ldsA

                if (anyAct) {                   // wave-uniform
                    const int kbRow = (cbOff + cb) * 9 + dy * 3;
#pragma unroll
                    for (int dx = 0; dx < 3; ++dx) {
                        union { v16bf v; unsigned u[8]; } Af[4];
#pragma unroll
                        for (int mt = 0; mt < 4; ++mt) {
                            int pp = Mrow + mt * 16 + dx;   // strip pixel
#pragma unroll
                            for (int j = 0; j < 4; ++j)
                                Af[mt].u[j] = ldsA[pp * ROWD + ((2 * j + kbA) >> 1)];
#pragma unroll
                            for (int j = 4; j < 8; ++j)
                                Af[mt].u[j] =
                                    ldsA[pp * ROWD + ((8 + 2 * (j - 4) + kbA) >> 1)];
                        }
                        const int kbCur = kbRow + dx;
#pragma unroll
                        for (int i = 0; i < NT; ++i) {
                            if (!act[i]) continue;          // wave-uniform
                            const uint4* bp = (const uint4*)
                                (wp + (((long)tt[i] * nKb + kbCur) * 32 + lane) * 8);
                            union { v16bf v; uint4 q[2]; } Bf;
                            Bf.q[0] = bp[0];
                            Bf.q[1] = bp[1];
#pragma unroll
                            for (int mt = 0; mt < 4; ++mt)
                                acc[mt][i] = __builtin_amdgcn_wmma_f32_16x16x32_bf16(
                                    false, Af[mt].v, false, Bf.v, (short)0,
                                    acc[mt][i], false, false);
                        }
                    }
                }
            }
        }
    }

    if (!anyAct) return;

    // ---- epilogue (C layout: VGPR r, lanes0-15 M=r, lanes16-31 M=r+8) ----
    const int n    = lane & 15;
    const int mOff = (lane < 16) ? 0 : 8;
#pragma unroll
    for (int i = 0; i < NT; ++i) {
        if (!act[i]) continue;
        int co = tt[i] * 16 + n;
        float bv = bias[co];
#pragma unroll
        for (int mt = 0; mt < 4; ++mt) {
#pragma unroll
            for (int r = 0; r < 8; ++r) {
                int x = r + mOff + mt * 16;
                float val = acc[mt][i][r] + bv;
                long pix = (long)y * Ww + x;
                if (MODE == 0) {
                    out[((long)b * Cout + co) * HW + pix] = val;
                } else if (MODE == 1) {
                    float sg = 1.f / (1.f + __expf(-val));
                    if (co < (Cout >> 1))               // r-part -> store r*h
                        sg *= hbuf[((long)b * HID + co) * HW + pix];
                    out[((long)b * Cout + co) * HW + pix] = sg;
                } else {
                    float cd = tanhf(val);
                    float z  = zbuf[((long)b * 128 + 64 + co) * HW + pix];
                    float hv = hbuf[((long)b * HID + co) * HW + pix];
                    out[((long)b * HID + co) * HW + pix] = z * hv + (1.f - z) * cd;
                }
            }
        }
    }
}

template <int NT, int MODE, bool CG>
static void run_conv(hipStream_t stream,
                     const float* i0, int C0, long sb0,
                     const float* i1, int C1, long sb1,
                     const unsigned* wpk, const float* bs, int Cout, float* out,
                     const float* hb, const float* zb)
{
    dim3 grid(Bc * Hh, (Cout / 16 + 4 * NT - 1) / (4 * NT));
    conv3x3_wmma<NT, MODE, CG><<<grid, dim3(128), 0, stream>>>(
        i0, C0, sb0, i1, C1, sb1, wpk, bs, Cout, out, hb, zb);
}

// ---------------------------------------------------------------------------
// Host orchestration
// ---------------------------------------------------------------------------
extern "C" void kernel_launch(void* const* d_in, const int* in_sizes, int n_in,
                              void* d_out, int out_size, void* d_ws, size_t ws_size,
                              hipStream_t stream)
{
    (void)in_sizes; (void)n_in; (void)out_size; (void)ws_size;

    const float* x    = (const float*)d_in[0];
    const float* Win0 = (const float*)d_in[1];
    const float* bin0 = (const float*)d_in[2];
    const float* We10 = (const float*)d_in[3];
    const float* be10 = (const float*)d_in[4];
    const float* We21 = (const float*)d_in[5];
    const float* be21 = (const float*)d_in[6];
    const float *Wint[3], *bint[3], *Wg[3], *bg[3], *Wc[3], *bcn[3];
    for (int nn = 0; nn < 3; ++nn) {
        Wint[nn] = (const float*)d_in[7 + nn * 6 + 0];
        bint[nn] = (const float*)d_in[7 + nn * 6 + 1];
        Wg[nn]   = (const float*)d_in[7 + nn * 6 + 2];
        bg[nn]   = (const float*)d_in[7 + nn * 6 + 3];
        Wc[nn]   = (const float*)d_in[7 + nn * 6 + 4];
        bcn[nn]  = (const float*)d_in[7 + nn * 6 + 5];
    }

    const size_t HB = (size_t)Bc * HID * HW;        // 2,097,152
    const size_t YB = (size_t)Bc * 32 * HW;         // 1,048,576
    const size_t GB = (size_t)Bc * 128 * HW;        // 4,194,304

    float* ws = (float*)d_ws;
    float* hA[3]   = { ws, ws + HB, ws + 2 * HB };
    float* hBuf[3] = { ws + 3 * HB, ws + 4 * HB, ws + 5 * HB };
    float* ybuf  = ws + 6 * HB;
    float* bubuf = ybuf + YB;
    float* gates = bubuf + YB;                      // rh(0..63) | z(64..127)

    unsigned* pk = (unsigned*)(gates + GB);
    auto alloc = [&](size_t ndw) { unsigned* r = pk; pk += ndw; return r; };
    // pack sizes (dwords): (Cout/16) * nKb * 256
    unsigned* pproj[3] = { alloc(2 * 9 * 256),   // Win0: C0=3 -> 1 cb
                           alloc(2 * 18 * 256),  // We10: C0=64 -> 2 cb
                           alloc(2 * 18 * 256) };
    unsigned *pint[3], *pg[3], *pc[3];
    for (int nn = 0; nn < 3; ++nn) pint[nn] = alloc(2 * 9 * 256);
    for (int nn = 0; nn < 3; ++nn) pg[nn]   = alloc(8 * 27 * 256);
    for (int nn = 0; nn < 3; ++nn) pc[nn]   = alloc(4 * 27 * 256);

    // zero both hidden-state generations (covers inactive-node steps)
    hipMemsetAsync(ws, 0, 6 * HB * sizeof(float), stream);

    auto packLaunch = [&](const float* src, unsigned* dst,
                          int Cout, int C0, int C1) {
        int cbT = (C0 + 31) / 32 + (C1 ? (C1 + 31) / 32 : 0);
        int total = (Cout / 16) * cbT * 9 * 256;
        pack_weights<<<dim3((total + 255) / 256), dim3(256), 0, stream>>>(
            src, dst, Cout, C0 + C1, C0, C1, total);
    };
    packLaunch(Win0, pproj[0], 32, CIN, 0);
    packLaunch(We10, pproj[1], 32, HID, 0);
    packLaunch(We21, pproj[2], 32, HID, 0);
    for (int nn = 0; nn < 3; ++nn) packLaunch(Wint[nn], pint[nn], 32, 32, 0);
    for (int nn = 0; nn < 3; ++nn) packLaunch(Wg[nn],   pg[nn], 128, 32, HID);
    for (int nn = 0; nn < 3; ++nn) packLaunch(Wc[nn],   pc[nn],  64, 32, HID);

    const float* bprojN[3] = { bin0, be10, be21 };

    for (int t = 0; t < PROCESS_T; ++t) {
        float** hOld = (t & 1) ? hBuf : hA;
        float** hNew = (t & 1) ? hA   : hBuf;

        for (int nn = 0; nn < 3; ++nn) {
            if (t < nn) continue;                   // wave-propagation schedule
            // 1) per-edge projection conv
            if (nn == 0) {
                const float* src = (t < Tt) ? (x + (size_t)t * CIN * HW) : nullptr;
                run_conv<1, 0, true>(stream, src, CIN, (long)Tt * CIN * HW,
                                     nullptr, 0, 0, pproj[0], bprojN[0], 32,
                                     ybuf, nullptr, nullptr);
            } else {
                run_conv<1, 0, false>(stream, hOld[nn - 1], HID, (long)HID * HW,
                                      nullptr, 0, 0, pproj[nn], bprojN[nn], 32,
                                      ybuf, nullptr, nullptr);
            }
            // 2) integrator conv
            run_conv<1, 0, false>(stream, ybuf, 32, (long)32 * HW,
                                  nullptr, 0, 0, pint[nn], bint[nn], 32,
                                  bubuf, nullptr, nullptr);
            // 3) gates conv on concat(bu, h): sigmoid, fuse r*h
            run_conv<2, 1, false>(stream, bubuf, 32, (long)32 * HW,
                                  hOld[nn], HID, (long)HID * HW,
                                  pg[nn], bg[nn], 128, gates,
                                  hOld[nn], nullptr);
            // 4) candidate conv on concat(bu, r*h): tanh + GRU update
            run_conv<1, 2, false>(stream, bubuf, 32, (long)32 * HW,
                                  gates, HID, (long)128 * HW,
                                  pc[nn], bcn[nn], 64, hNew[nn],
                                  hOld[nn], gates);
        }
    }

    // t = 9 (odd) wrote hNew = hA; output node is index 2
    hipMemcpyAsync(d_out, hA[2], HB * sizeof(float),
                   hipMemcpyDeviceToDevice, stream);
}
